// SwinLSTM_8065948582604
// MI455X (gfx1250) — matrix operand order repaired and checked by reading, not compiled, not run on hardware
//
#include <hip/hip_runtime.h>
#include <hip/hip_bf16.h>
#include <math.h>

#define Bn 16
#define Cn 192
#define Hn 64
#define Wn 64
#define HWn 4096
#define CPG 48                 // channels per group
#define NGRP (Bn * 4)          // 64 (b, group) pairs
#define GSIZE (CPG * HWn)      // 196608 elements per group-norm group
#define BSTRIDE 200            // padded LDS row stride (halfs) to avoid bank conflicts

typedef __attribute__((ext_vector_type(16))) _Float16 v16h;
typedef __attribute__((ext_vector_type(8)))  _Float16 v8h;
typedef __attribute__((ext_vector_type(8)))  float    v8f;

__device__ __forceinline__ float sigm(float x) { return 1.0f / (1.0f + expf(-x)); }

// ---------------------------------------------------------------------------
// Kernel 0: convert pointwise weights fp32 -> f16 ([4C, C], C contiguous)
// ---------------------------------------------------------------------------
__global__ void prep_kernel(const float* __restrict__ pw, _Float16* __restrict__ wb) {
    int i = blockIdx.x * 256 + threadIdx.x;
    if (i < 4 * Cn * Cn) wb[i] = (_Float16)pw[i];
}

// ---------------------------------------------------------------------------
// Kernel 1: z2 = dwconv7x7(x + hidden) + dw_b ; deterministic partial sums
// block (16,16), grid (4, 4, B*C). LDS tile 22x22 with halo=3, zero padded.
// ---------------------------------------------------------------------------
__global__ void dwconv_kernel(const float* __restrict__ x, const float* __restrict__ hid,
                              const float* __restrict__ dww, const float* __restrict__ dwb,
                              float* __restrict__ z2,
                              float* __restrict__ partS, float* __restrict__ partQ) {
    __shared__ float tile[22][22];
    __shared__ float wsm[49];
    __shared__ float red[8][2];

    int bc = blockIdx.z;
    int b  = bc / Cn;
    int c  = bc % Cn;
    int h0 = blockIdx.y * 16, w0 = blockIdx.x * 16;
    int tid = threadIdx.y * 16 + threadIdx.x;

    const float* xp = x   + (size_t)bc * HWn;
    const float* hp = hid + (size_t)bc * HWn;

    if (tid < 49) wsm[tid] = dww[c * 49 + tid];

    for (int idx = tid; idx < 22 * 22; idx += 256) {
        int ly = idx / 22, lx = idx % 22;
        int gy = h0 + ly - 3, gx = w0 + lx - 3;
        float v = 0.0f;
        if (gy >= 0 && gy < Hn && gx >= 0 && gx < Wn) {
            int o = gy * Wn + gx;
            v = xp[o] + hp[o];
        }
        tile[ly][lx] = v;
    }
    __syncthreads();

    float acc = dwb[c];
#pragma unroll
    for (int ky = 0; ky < 7; ++ky)
#pragma unroll
        for (int kx = 0; kx < 7; ++kx)
            acc = fmaf(tile[threadIdx.y + ky][threadIdx.x + kx], wsm[ky * 7 + kx], acc);

    z2[(size_t)bc * HWn + (h0 + threadIdx.y) * Wn + (w0 + threadIdx.x)] = acc;

    // deterministic block reduction of sum / sumsq
    float s = acc, q = acc * acc;
#pragma unroll
    for (int off = 16; off > 0; off >>= 1) {
        s += __shfl_down(s, off, 32);
        q += __shfl_down(q, off, 32);
    }
    int wave = tid >> 5, lane = tid & 31;
    if (lane == 0) { red[wave][0] = s; red[wave][1] = q; }
    __syncthreads();
    if (tid == 0) {
        float S = 0.0f, Q = 0.0f;
#pragma unroll
        for (int i = 0; i < 8; ++i) { S += red[i][0]; Q += red[i][1]; }
        int g = c / CPG;
        int tileId = blockIdx.y * 4 + blockIdx.x;              // 0..15
        int gi = (b * 4 + g) * 768 + (c % CPG) * 16 + tileId;  // 768 partials per group
        partS[gi] = S;
        partQ[gi] = Q;
    }
}

// ---------------------------------------------------------------------------
// Kernel 2: reduce 768 partials per (b, group) -> mean, rstd (fixed order)
// ---------------------------------------------------------------------------
__global__ void gstats_kernel(const float* __restrict__ partS, const float* __restrict__ partQ,
                              float* __restrict__ stats) {
    __shared__ float rs[256], rq[256];
    int g = blockIdx.x;
    int t = threadIdx.x;
    const float* ps = partS + g * 768;
    const float* pq = partQ + g * 768;
    rs[t] = ps[t] + ps[t + 256] + ps[t + 512];
    rq[t] = pq[t] + pq[t + 256] + pq[t + 512];
    __syncthreads();
    for (int off = 128; off > 0; off >>= 1) {
        if (t < off) { rs[t] += rs[t + off]; rq[t] += rq[t + off]; }
        __syncthreads();
    }
    if (t == 0) {
        float mean = rs[0] / (float)GSIZE;
        float var  = rq[0] / (float)GSIZE - mean * mean;
        stats[g * 2]     = mean;
        stats[g * 2 + 1] = rsqrtf(var + 1e-5f);
    }
}

// ---------------------------------------------------------------------------
// Kernel 3: GroupNorm + FiLM, fp32 -> f16, NCHW -> [B*H*W, C] via LDS transpose
// block (32, 8), grid (HW/32, C/32, B)
// ---------------------------------------------------------------------------
__global__ void norm_kernel(const float* __restrict__ z2, const float* __restrict__ ctx,
                            const float* __restrict__ stats, _Float16* __restrict__ zt) {
    __shared__ _Float16 t[32][33];
    int b   = blockIdx.z;
    int c0  = blockIdx.y * 32;
    int hw0 = blockIdx.x * 32;
    int tx  = threadIdx.x, ty = threadIdx.y;
#pragma unroll
    for (int j = 0; j < 4; ++j) {
        int cc = ty + j * 8;
        int c  = c0 + cc;
        int g  = c / CPG;
        float mean = stats[(b * 4 + g) * 2];
        float rstd = stats[(b * 4 + g) * 2 + 1];
        float z  = z2[((size_t)(b * Cn + c)) * HWn + hw0 + tx];
        float sc = ctx[((size_t)(b * 2 * Cn + c)) * HWn + hw0 + tx];
        float bi = ctx[((size_t)(b * 2 * Cn + Cn + c)) * HWn + hw0 + tx];
        float v = (z - mean) * rstd * (1.0f + sc) + bi;
        t[cc][tx] = (_Float16)v;
    }
    __syncthreads();
#pragma unroll
    for (int j = 0; j < 4; ++j) {
        int row = ty + j * 8;
        zt[((size_t)(b * HWn + hw0 + row)) * Cn + c0 + tx] = t[tx][row];
    }
}

// ---------------------------------------------------------------------------
// Kernel 4: WMMA GEMM (M=65536, N=768, K=192) fused with LSTM epilogue.
// - B (64 x 192 weight block, same for all waves in WG) staged in LDS once,
//   read via ds_load_b128 (DScnt) overlapping A's global_load_b128 (LOADcnt).
// - Each wave computes TWO adjacent 16x16 M-tiles per channel block: every B
//   fragment feeds 2 WMMAs (48 WMMAs/wave), then fused LSTM gate epilogue.
// block 256 (8 waves), grid (256, 12).
// ---------------------------------------------------------------------------
__global__ void __launch_bounds__(256) gemm_lstm_kernel(
        const _Float16* __restrict__ zt, const _Float16* __restrict__ wb,
        const float* __restrict__ pwb, const float* __restrict__ cell,
        float* __restrict__ out) {
    __shared__ _Float16 Bs[64 * BSTRIDE];   // 64 rows x 192 halfs, padded to 200

    int tid  = threadIdx.x;
    int wave = tid >> 5;
    int lane = tid & 31;
    int cblk = blockIdx.y;              // channel block 0..11
    int half = lane >> 4;               // 0 or 1
    int nloc = lane & 15;

    // ---- stage B block into LDS: 64 rows x 24 chunks of 8 halfs = 1536 chunks
#pragma unroll
    for (int j = 0; j < 6; ++j) {
        int chunk = tid + j * 256;
        int row   = chunk / 24;          // 0..63  (gate*16 + n)
        int col   = (chunk % 24) * 8;    // 0..184
        int gate  = row >> 4, nl = row & 15;
        v8h v = *(const v8h*)(wb + ((size_t)(gate * Cn + cblk * 16 + nl)) * Cn + col);
        *(v8h*)(&Bs[row * BSTRIDE + col]) = v;
    }
    __syncthreads();

    int mt0 = (blockIdx.x * 8 + wave) * 2;   // first of two adjacent M-tiles

    v8f acc[2][4] = {};

    // A fragment: lane holds row M = lane%16, K offset 0/8 by half (ISA 7.12.2).
    // Second tile is rows +16 => +16*192 halfs from same base.
    const _Float16* Arow = zt + ((size_t)(mt0 * 16 + nloc)) * Cn + half * 8;

#pragma unroll
    for (int ks = 0; ks < 6; ++ks) {          // K = 192 = 6 * 32
        union { v16h v; v8h h[2]; } a0, a1;
        a0.h[0] = *(const v8h*)(Arow + ks * 32);
        a0.h[1] = *(const v8h*)(Arow + ks * 32 + 16);
        a1.h[0] = *(const v8h*)(Arow + 16 * Cn + ks * 32);
        a1.h[1] = *(const v8h*)(Arow + 16 * Cn + ks * 32 + 16);
#pragma unroll
        for (int g = 0; g < 4; ++g) {
            union { v16h v; v8h h[2]; } bm;
            const _Float16* bp = &Bs[(g * 16 + nloc) * BSTRIDE + ks * 32 + half * 8];
            bm.h[0] = *(const v8h*)(bp);
            bm.h[1] = *(const v8h*)(bp + 16);
            acc[0][g] = __builtin_amdgcn_wmma_f32_16x16x32_f16(
                false, a0.v, false, bm.v, (short)0, acc[0][g], false, false);
            acc[1][g] = __builtin_amdgcn_wmma_f32_16x16x32_f16(
                false, a1.v, false, bm.v, (short)0, acc[1][g], false, false);
        }
    }

    // Epilogue: C/D layout -> VGPR r holds M = r + 8*half, lane%16 holds N
    int c  = cblk * 16 + nloc;
    float bF = pwb[c], bI = pwb[Cn + c], bG = pwb[2 * Cn + c], bO = pwb[3 * Cn + c];
#pragma unroll
    for (int t = 0; t < 2; ++t) {
        int mbase = (mt0 + t) * 16 + half * 8;
        int b  = mbase >> 12;       // 4096 spatial positions per image
        int hw = mbase & 4095;
        size_t cidx = ((size_t)(b * Cn + c)) * HWn + hw;
        const float* cp = cell + cidx;
        float* ho = out + cidx;                                 // hidden output
        float* co = out + (size_t)Bn * Cn * HWn + cidx;         // cell output
#pragma unroll
        for (int r = 0; r < 8; ++r) {
            float f = acc[t][0][r] + bF;
            float i = acc[t][1][r] + bI;
            float g = acc[t][2][r] + bG;
            float o = acc[t][3][r] + bO;
            float cold = cp[r];
            float cnew = sigm(f) * cold + sigm(i) * tanhf(g);
            float gel  = 0.5f * cnew * (1.0f + erff(cnew * 0.70710678118654752f));
            ho[r] = sigm(o) * gel;
            co[r] = cnew;
        }
    }
}

// ---------------------------------------------------------------------------
extern "C" void kernel_launch(void* const* d_in, const int* in_sizes, int n_in,
                              void* d_out, int out_size, void* d_ws, size_t ws_size,
                              hipStream_t stream) {
    const float* x      = (const float*)d_in[0];
    const float* hidden = (const float*)d_in[1];
    const float* cell   = (const float*)d_in[2];
    const float* ctx    = (const float*)d_in[3];
    const float* dww    = (const float*)d_in[4];
    const float* dwb    = (const float*)d_in[5];
    const float* pww    = (const float*)d_in[6];
    const float* pwb    = (const float*)d_in[7];
    float* out = (float*)d_out;

    char* ws = (char*)d_ws;
    size_t off = 0;
    _Float16* Wb = (_Float16*)(ws + off); off += (size_t)4 * Cn * Cn * sizeof(_Float16);   // 288 KB
    float*    z2 = (float*)(ws + off);    off += (size_t)Bn * Cn * HWn * sizeof(float);    // 48 MB
    _Float16* zt = (_Float16*)(ws + off); off += (size_t)Bn * HWn * Cn * sizeof(_Float16); // 24 MB
    float* partS = (float*)(ws + off);    off += (size_t)NGRP * 768 * sizeof(float);
    float* partQ = (float*)(ws + off);    off += (size_t)NGRP * 768 * sizeof(float);
    float* stats = (float*)(ws + off);    off += (size_t)NGRP * 2 * sizeof(float);

    prep_kernel<<<(4 * Cn * Cn + 255) / 256, 256, 0, stream>>>(pww, Wb);
    dwconv_kernel<<<dim3(4, 4, Bn * Cn), dim3(16, 16), 0, stream>>>(
        x, hidden, dww, dwb, z2, partS, partQ);
    gstats_kernel<<<NGRP, 256, 0, stream>>>(partS, partQ, stats);
    norm_kernel<<<dim3(HWn / 32, Cn / 32, Bn), dim3(32, 8), 0, stream>>>(z2, ctx, stats, zt);
    gemm_lstm_kernel<<<dim3((Bn * HWn / 16) / 16, Cn / 16), 256, 0, stream>>>(
        zt, Wb, pwb, cell, out);
}
